// GroupShuffleAttention_2173253452291
// MI455X (gfx1250) — matrix operand order repaired and checked by reading, not compiled, not run on hardware
//
#include <hip/hip_runtime.h>

typedef __attribute__((ext_vector_type(16))) _Float16     v16h;
typedef __attribute__((ext_vector_type(2)))  __fp16       v2fp;   // cvt_pkrtz result type
typedef __attribute__((ext_vector_type(8)))  float        v8f;
typedef __attribute__((ext_vector_type(4)))  unsigned int v4u;

#define B_   4
#define C_   512
#define N_   2048
#define G_   8
#define CG_  64
#define NTOT_ (B_*C_*N_)

// reinterpret helper between 2x16B loads and a WMMA 16xf16 fragment
union FragAB { v4u q[2]; v16h v; _Float16 h[16]; };

// lanes 0-15 <-> 16-31 swap as pure VALU (v_permlanex16_b32, identity selects)
__device__ __forceinline__ int swap16i(int x) {
  return __builtin_amdgcn_permlanex16(0, x, 0x76543210, 0xfedcba98, false, false);
}
__device__ __forceinline__ float swap16f(float x) {
  return __int_as_float(swap16i(__float_as_int(x)));
}

// ---------------------------------------------------------------------------
// Pass 1: grouped 1x1 conv (f32), write t^T as f16 [b,g,n,c] and elu(t) as
// f16 [b,g,c,n] into workspace. ~1 GFLOP + 17MB r / 16MB w: negligible.
// ---------------------------------------------------------------------------
__global__ __launch_bounds__(256) void conv_elu_kernel(
    const float* __restrict__ points, const float* __restrict__ conv_w,
    const float* __restrict__ conv_b, _Float16* __restrict__ tT,
    _Float16* __restrict__ vv)
{
  __shared__ float sW[64*64];
  __shared__ float sX[64*64];
  __shared__ float sT[64*65];           // padded to avoid bank conflicts on transpose
  const int blk   = blockIdx.x;
  const int ntile = blk & 31;           // N_/64 = 32 tiles
  const int bg    = blk >> 5;
  const int g = bg & 7, b = bg >> 3;
  const int n0  = ntile * 64;
  const int tid = threadIdx.x;

  for (int i = tid; i < 64*64; i += 256) {
    sW[i] = conv_w[g*64*64 + i];                     // [o][ci]
    const int ci = i >> 6, n = i & 63;
    sX[i] = points[((size_t)b*C_ + g*64 + ci)*N_ + n0 + n];   // [ci][n]
  }
  __syncthreads();

  const int o  = tid & 63;
  const int nq = tid >> 6;              // 0..3 -> 16 columns each
  float acc[16];
  const float bias = conv_b[g*64 + o];
  #pragma unroll
  for (int j = 0; j < 16; ++j) acc[j] = bias;
  for (int ci = 0; ci < 64; ++ci) {
    const float w = sW[o*64 + ci];
    #pragma unroll
    for (int j = 0; j < 16; ++j) acc[j] += w * sX[ci*64 + nq*16 + j];
  }
  #pragma unroll
  for (int j = 0; j < 16; ++j) sT[o*65 + nq*16 + j] = acc[j];
  __syncthreads();

  {   // v = elu(t), layout [b,g,cg,n], 32B per thread
    const int cg = tid >> 2, ch = tid & 3;
    FragAB u;
    #pragma unroll
    for (int j = 0; j < 16; ++j) {
      const float t = sT[cg*65 + ch*16 + j];
      const float e = t > 0.f ? t : (__expf(t) - 1.f);
      u.h[j] = (_Float16)e;
    }
    v4u* dst = (v4u*)(vv + ((size_t)(b*G_ + g)*64 + cg)*N_ + n0 + ch*16);
    dst[0] = u.q[0]; dst[1] = u.q[1];
  }
  {   // t^T, layout [b,g,n,c], 32B per thread
    const int n = tid >> 2, ch = tid & 3;
    FragAB u;
    #pragma unroll
    for (int j = 0; j < 16; ++j) u.h[j] = (_Float16)sT[(ch*16 + j)*65 + n];
    v4u* dst = (v4u*)(tT + ((size_t)(b*G_ + g)*N_ + n0 + n)*64 + ch*16);
    dst[0] = u.q[0]; dst[1] = u.q[1];
  }
}

// ---------------------------------------------------------------------------
// Pass 2: fused flash attention (softmax over the key/query axis n per output
// column m). One block = (b,g) x 128-column tile; 8 waves x 16 columns.
// Per 32-key block: 4 wmma for s = t^T t, online softmax, 4 wmma for elu(t)@P.
// Writes (attn_out + identity) in channel-shuffled order straight to d_out.
// ---------------------------------------------------------------------------
__global__ __launch_bounds__(256) void flash_attn_kernel(
    const _Float16* __restrict__ tT, const _Float16* __restrict__ vv,
    const float* __restrict__ points, float* __restrict__ y)
{
  __shared__ __align__(16) _Float16 sK[32*72];   // keys tile  [n][c], padded
  __shared__ __align__(16) _Float16 sV[64*40];   // value tile [c][n], padded
  const int blk   = blockIdx.x;
  const int mtile = blk & 15;                    // N_/128 = 16
  const int bg    = blk >> 4;
  const int g = bg & 7, b = bg >> 3;
  const int tid  = threadIdx.x;
  const int wave = tid >> 5;
  const int lane = tid & 31;
  const int lh   = lane >> 4;                    // half-lane group
  const int lm   = lane & 15;
  const int m    = mtile*128 + wave*16 + lm;     // this lane's output column

  const _Float16* tTg = tT + (size_t)(b*G_ + g) * N_ * 64;
  const _Float16* vvg = vv + (size_t)(b*G_ + g) * 64 * N_;

  // B fragments of s = A(n,c) x B(c,m): per lane N=m=lm, element i holds
  // K = 16*lh + i (+32 for chunk 1). Constant across the whole n loop.
  // Fold the softmax scale 1/sqrt(CG)=0.125 (exact in f16) into B once.
  FragAB bm0, bm1;
  {
    const v4u* p0 = (const v4u*)(tTg + (size_t)m*64 + 16*lh);
    bm0.q[0] = p0[0]; bm0.q[1] = p0[1];
    const v4u* p1 = (const v4u*)(tTg + (size_t)m*64 + 32 + 16*lh);
    bm1.q[0] = p1[0]; bm1.q[1] = p1[1];
    #pragma unroll
    for (int i = 0; i < 16; ++i) {
      bm0.h[i] = (_Float16)(bm0.h[i] * (_Float16)0.125f);
      bm1.h[i] = (_Float16)(bm1.h[i] * (_Float16)0.125f);
    }
  }

  v8f acc[4] = {};                   // out[c,m]: 4 c-tiles of 16, column m
  float mx = -1e30f, l = 0.f;        // running max / denom for column m

  const int krow = tid >> 3, kch = tid & 7;   // sK fill: 32 rows x 8 x 16B
  const int vrow = tid >> 2, vch = tid & 3;   // sV fill: 64 rows x 4 x 16B

  for (int n0 = 0; n0 < N_; n0 += 32) {
    // cooperative global->LDS staging (b128 each way)
    const v4u dk = *(const v4u*)(tTg + (size_t)(n0 + krow)*64 + kch*8);
    const v4u dv = *(const v4u*)(vvg + (size_t)vrow*N_ + n0 + vch*8);
    *(v4u*)(sK + krow*72 + kch*8) = dk;
    *(v4u*)(sV + vrow*40 + vch*8) = dv;
    __syncthreads();

    // A fragments: M = row (lm), element i holds K = 16*lh + i
    FragAB a00, a01, a10, a11, av0, av1, av2, av3;
    {
      const _Float16* r0 = sK + lm*72 + 16*lh;        // n-subtile 0
      a00.q[0] = *(const v4u*)(r0);       a00.q[1] = *(const v4u*)(r0 + 8);
      a01.q[0] = *(const v4u*)(r0 + 32);  a01.q[1] = *(const v4u*)(r0 + 40);
      const _Float16* r1 = sK + (lm + 16)*72 + 16*lh; // n-subtile 1
      a10.q[0] = *(const v4u*)(r1);       a10.q[1] = *(const v4u*)(r1 + 8);
      a11.q[0] = *(const v4u*)(r1 + 32);  a11.q[1] = *(const v4u*)(r1 + 40);
      const _Float16* rv = sV + lm*40 + 16*lh;        // v[c-tile][n-block]
      av0.q[0] = *(const v4u*)(rv);          av0.q[1] = *(const v4u*)(rv + 8);
      av1.q[0] = *(const v4u*)(rv + 16*40);  av1.q[1] = *(const v4u*)(rv + 16*40 + 8);
      av2.q[0] = *(const v4u*)(rv + 32*40);  av2.q[1] = *(const v4u*)(rv + 32*40 + 8);
      av3.q[0] = *(const v4u*)(rv + 48*40);  av3.q[1] = *(const v4u*)(rv + 48*40 + 8);
    }
    __syncthreads();

    // s[n-block, m] = (t^T t)/8  (K = c = 64, two chained K=32 WMMAs/subtile)
    v8f sf0 = {}, sf1 = {};
    sf0 = __builtin_amdgcn_wmma_f32_16x16x32_f16(false, a00.v, false, bm0.v, (short)0, sf0, false, false);
    sf0 = __builtin_amdgcn_wmma_f32_16x16x32_f16(false, a01.v, false, bm1.v, (short)0, sf0, false, false);
    sf1 = __builtin_amdgcn_wmma_f32_16x16x32_f16(false, a10.v, false, bm0.v, (short)0, sf1, false, false);
    sf1 = __builtin_amdgcn_wmma_f32_16x16x32_f16(false, a11.v, false, bm1.v, (short)0, sf1, false, false);

    // online softmax over n for column m (lane-local + one half-lane swap)
    float p0[8], p1[8];
    float bmax = -1e30f;
    #pragma unroll
    for (int r = 0; r < 8; ++r) {
      p0[r] = sf0[r];
      p1[r] = sf1[r];
      bmax = fmaxf(bmax, fmaxf(p0[r], p1[r]));
    }
    bmax = fmaxf(bmax, swap16f(bmax));
    // uniform skip of the rescale when no lane's max moved (steady state)
    if (__ballot(bmax > mx)) {
      const float newmx = fmaxf(mx, bmax);
      const float fac   = __expf(mx - newmx);
      l *= fac;
      #pragma unroll
      for (int i = 0; i < 4; ++i)
        #pragma unroll
        for (int r = 0; r < 8; ++r) acc[i][r] *= fac;
      mx = newmx;
    }
    float psum = 0.f;
    #pragma unroll
    for (int r = 0; r < 8; ++r) {
      p0[r] = __expf(p0[r] - mx);
      p1[r] = __expf(p1[r] - mx);
      psum += p0[r] + p1[r];
    }
    l += psum + swap16f(psum);

    // re-layout P (C/D layout) -> B fragment layout: element i needs
    // n = 16*lh + i. Pack to f16x2 first, then a single swap per dword:
    // lane sends the fragment its partner needs (lh=0 sends p1, lh=1 sends p0).
    union PBu { int u[8]; v16h v; } pb;
    #pragma unroll
    for (int q = 0; q < 4; ++q) {
      union H2u { v2fp h2; int u; } own0, own1;
      own0.h2 = __builtin_amdgcn_cvt_pkrtz(p0[2*q], p0[2*q + 1]);
      own1.h2 = __builtin_amdgcn_cvt_pkrtz(p1[2*q], p1[2*q + 1]);
      const int send = lh ? own0.u : own1.u;
      const int recv = swap16i(send);
      pb.u[q]     = lh ? recv   : own0.u;   // n = 16*lh + {0..7}
      pb.u[4 + q] = lh ? own1.u : recv;     // n = 16*lh + {8..15}
    }

    acc[0] = __builtin_amdgcn_wmma_f32_16x16x32_f16(false, av0.v, false, pb.v, (short)0, acc[0], false, false);
    acc[1] = __builtin_amdgcn_wmma_f32_16x16x32_f16(false, av1.v, false, pb.v, (short)0, acc[1], false, false);
    acc[2] = __builtin_amdgcn_wmma_f32_16x16x32_f16(false, av2.v, false, pb.v, (short)0, acc[2], false, false);
    acc[3] = __builtin_amdgcn_wmma_f32_16x16x32_f16(false, av3.v, false, pb.v, (short)0, acc[3], false, false);
  }

  // finalize: divide by softmax denom, channel-shuffle, add identity, store
  const float invl = 1.f / l;
  #pragma unroll
  for (int ct = 0; ct < 4; ++ct) {
    #pragma unroll
    for (int r = 0; r < 8; ++r) {
      const int cg   = r + 8*lh + 16*ct;          // C/D layout row
      const int cnew = cg*G_ + g;                 // channel shuffle
      const size_t off = ((size_t)b*C_ + cnew)*N_ + m;
      y[off] = acc[ct][r] * invl + points[off];
    }
  }
}

// ---------------------------------------------------------------------------
// Pass 3a: deterministic GroupNorm statistics. One block per (b, group):
// channels [64j, 64j+64) x N are contiguous in y.
// ---------------------------------------------------------------------------
__global__ __launch_bounds__(256) void gn_stats_kernel(
    const float* __restrict__ y, float* __restrict__ stats)
{
  __shared__ float rs[256], rq[256];
  const int bj = blockIdx.x;                      // b*G_ + j
  const float* base = y + (size_t)bj * CG_ * N_;
  float s = 0.f, q = 0.f;
  for (int i = threadIdx.x; i < CG_*N_; i += 256) {
    const float v = base[i];
    s += v; q += v*v;
  }
  rs[threadIdx.x] = s; rq[threadIdx.x] = q;
  __syncthreads();
  for (int d = 128; d > 0; d >>= 1) {
    if (threadIdx.x < d) {
      rs[threadIdx.x] += rs[threadIdx.x + d];
      rq[threadIdx.x] += rq[threadIdx.x + d];
    }
    __syncthreads();
  }
  if (threadIdx.x == 0) { stats[bj*2] = rs[0]; stats[bj*2 + 1] = rq[0]; }
}

// ---------------------------------------------------------------------------
// Pass 3b: normalize in place.
// ---------------------------------------------------------------------------
__global__ __launch_bounds__(256) void gn_apply_kernel(
    float* __restrict__ y, const float* __restrict__ stats,
    const float* __restrict__ gn_w, const float* __restrict__ gn_b)
{
  const size_t idx = (size_t)blockIdx.x * 256 + threadIdx.x;
  if (idx >= (size_t)NTOT_) return;
  const int b = (int)(idx / ((size_t)C_ * N_));
  const int c = (int)((idx / N_) % C_);
  const int j = c / CG_;
  const float cnt  = (float)(CG_ * N_);
  const float mean = stats[(b*G_ + j)*2] / cnt;
  const float var  = stats[(b*G_ + j)*2 + 1] / cnt - mean*mean;
  const float r    = rsqrtf(var + 1e-5f);
  y[idx] = (y[idx] - mean) * r * gn_w[c] + gn_b[c];
}

// ---------------------------------------------------------------------------
extern "C" void kernel_launch(void* const* d_in, const int* in_sizes, int n_in,
                              void* d_out, int out_size, void* d_ws, size_t ws_size,
                              hipStream_t stream) {
  (void)in_sizes; (void)n_in; (void)out_size; (void)ws_size;
  const float* points = (const float*)d_in[0];
  const float* conv_w = (const float*)d_in[1];
  const float* conv_b = (const float*)d_in[2];
  const float* gn_w   = (const float*)d_in[3];
  const float* gn_b   = (const float*)d_in[4];
  float* out = (float*)d_out;

  char* ws = (char*)d_ws;
  const size_t HALF16 = (size_t)B_ * G_ * N_ * CG_ * sizeof(_Float16);  // 8 MiB
  _Float16* tT    = (_Float16*)ws;                 // [b,g,n,c]
  _Float16* vv    = (_Float16*)(ws + HALF16);      // [b,g,c,n]
  float*    stats = (float*)(ws + 2*HALF16);       // [b,g][sum,sumsq]

  conv_elu_kernel<<<B_*G_*(N_/64), 256, 0, stream>>>(points, conv_w, conv_b, tT, vv);
  flash_attn_kernel<<<B_*G_*(N_/128), 256, 0, stream>>>(tT, vv, points, out);
  gn_stats_kernel<<<B_*G_, 256, 0, stream>>>(out, stats);
  gn_apply_kernel<<<(NTOT_ + 255)/256, 256, 0, stream>>>(out, stats, gn_w, gn_b);
}
